// MPNN_75333726372238
// MI455X (gfx1250) — compile-verified, matrix operand order checked by reading, not run on hardware
//
#include <hip/hip_runtime.h>

#define NN_NODES 50000
#define NN_EDGES 800000
#define NN_GRAPHS 256
#define IN_CH 32
#define PE_DIM 24
#define HID 64
#define NLAYERS 3

typedef __attribute__((ext_vector_type(16))) _Float16 v16h;
typedef __attribute__((ext_vector_type(8)))  float    v8f;

// SiLU via CDNA5 hardware tanh (TRANS), with the 0.5 pre-scale already folded
// into the operand:  given hx = 0.5*z, silu(z) = hx + hx*tanh(hx).
// v_nop inside the asm satisfies the TRANS result-use hazard (ISA 5.7.4);
// the fma + f16-convert fuse into v_fma_mixlo_f16 at the stores.
__device__ __forceinline__ float silu_from_half(float hx) {
    float t;
    asm("v_tanh_f32 %0, %1\n\tv_nop" : "=v"(t) : "v"(hx));
    return __builtin_fmaf(hx, t, hx);
}

__device__ __forceinline__ float silu_f(float x) {
    return silu_from_half(0.5f * x);
}

// B operand (32x16 f16) for V_WMMA_F32_16X16X32_F16, pre-scaled by `s`.
// lane L holds column n = L%16; K = kt*32 + (L/16)*16 + j, j = 0..15
__device__ __forceinline__ v16h load_B_tile(const float* __restrict__ W,
                                            int kt, int nt, int lane, float s) {
    int n  = nt * 16 + (lane & 15);
    int k0 = kt * 32 + ((lane >> 4) << 4);
    v16h b;
#pragma unroll
    for (int j = 0; j < 16; ++j)
        b[j] = (_Float16)(W[(k0 + j) * HID + n] * s);
    return b;
}

// A operand (16x32 f16): lane L holds row m = L%16, 16 consecutive K starting
// at c0 (c0 = 0 for lanes 0-15, 16 for lanes 16-31; +32 for the second K-step)
__device__ __forceinline__ v16h gather_A_tile(const float* __restrict__ row, int c0) {
    const float4* p = reinterpret_cast<const float4*>(row + c0);
    v16h a;
#pragma unroll
    for (int q = 0; q < 4; ++q) {
        float4 f = p[q];
        a[4*q+0] = (_Float16)f.x;
        a[4*q+1] = (_Float16)f.y;
        a[4*q+2] = (_Float16)f.z;
        a[4*q+3] = (_Float16)f.w;
    }
    return a;
}

// Fused 2-layer MLP over 16-row tiles with gather + scatter-add.
//   out[scatter[i]] += act2( silu( feat[gather[i]] @ W1 + b1 ) @ W2 + b2 )
// gatherIdx == nullptr -> identity gather (node rows).
// Stage-1 weights/bias are pre-scaled by 0.5 (folded SiLU); stage-2 likewise
// iff silu2, so the tanh input needs no separate multiply.
__global__ void __launch_bounds__(256)
k_mlp2_scatter(const float* __restrict__ feat,
               const int*   __restrict__ gatherIdx,
               const int*   __restrict__ scatterIdx,
               const float* __restrict__ W1, const float* __restrict__ b1,
               const float* __restrict__ W2, const float* __restrict__ b2,
               float* __restrict__ outAccum,
               int nTiles, int silu2)
{
    __shared__ __align__(32) _Float16 smem[8 * 16 * HID];   // 2 KB per wave

    const int lane   = threadIdx.x & 31;
    const int waveIB = threadIdx.x >> 5;
    const int waveG  = blockIdx.x * (blockDim.x >> 5) + waveIB;
    const int nWaves = gridDim.x * (blockDim.x >> 5);

    const float s2 = silu2 ? 0.5f : 1.0f;

    // Per-wave weight preload into VGPRs (f16), amortized over many tiles.
    // Stage-1 copies carry the folded 0.5 (exact: exponent shift).
    v16h B1r[2][4], B2r[2][4];
    float bia1[4], bia2[4];
#pragma unroll
    for (int kt = 0; kt < 2; ++kt)
#pragma unroll
        for (int nt = 0; nt < 4; ++nt) {
            B1r[kt][nt] = load_B_tile(W1, kt, nt, lane, 0.5f);
            B2r[kt][nt] = load_B_tile(W2, kt, nt, lane, s2);
        }
#pragma unroll
    for (int nt = 0; nt < 4; ++nt) {
        bia1[nt] = 0.5f * b1[nt * 16 + (lane & 15)];
        bia2[nt] = s2   * b2[nt * 16 + (lane & 15)];
    }

    const int col     = lane & 15;          // C-layout column / A-layout row
    const int c0      = (lane >> 4) << 4;   // 0 or 16 (A-layout K offset)
    const int halfOff = (lane >> 4) << 3;   // 0 or 8  (C-layout M offset)
    _Float16* myLds = smem + waveIB * (16 * HID);

    for (int tile = waveG; tile < nTiles; tile += nWaves) {
        // prefetch next iteration's index cachelines (global_prefetch)
        if (tile + nWaves < nTiles) {
            __builtin_prefetch(scatterIdx + (size_t)(tile + nWaves) * 16, 0, 0);
            if (gatherIdx)
                __builtin_prefetch(gatherIdx + (size_t)(tile + nWaves) * 16, 0, 0);
        }

        const int e   = tile * 16 + col;
        const int src = gatherIdx ? gatherIdx[e] : e;
        const float* hr = feat + (size_t)src * HID;

        v16h A0 = gather_A_tile(hr, c0);
        v16h A1 = gather_A_tile(hr, c0 + 32);

        // ---- stage 1: hx = 0.5*(A@W1+b1) via WMMA; silu = hx + hx*tanh(hx)
#pragma unroll
        for (int nt = 0; nt < 4; ++nt) {
            v8f acc;
#pragma unroll
            for (int v = 0; v < 8; ++v) acc[v] = bia1[nt];
            acc = __builtin_amdgcn_wmma_f32_16x16x32_f16(
                      false, A0, false, B1r[0][nt], (short)0, acc, false, false);
            acc = __builtin_amdgcn_wmma_f32_16x16x32_f16(
                      false, A1, false, B1r[1][nt], (short)0, acc, false, false);
#pragma unroll
            for (int v = 0; v < 8; ++v)
                myLds[(halfOff + v) * HID + nt * 16 + col] =
                    (_Float16)silu_from_half(acc[v]);
        }

        // ---- stage 2 A operands read back in A layout (same-wave DS, in-order)
        const _Float16* rp = myLds + col * HID + c0;
        v16h A2a = *reinterpret_cast<const v16h*>(rp);
        v16h A2b = *reinterpret_cast<const v16h*>(rp + 32);

        // 8 scatter destinations for this lane-half (loaded late: short live range)
        const int4* dp = reinterpret_cast<const int4*>(scatterIdx + tile * 16 + halfOff);
        int4 dlo = dp[0];
        int4 dhi = dp[1];
        int dsts[8] = { dlo.x, dlo.y, dlo.z, dlo.w, dhi.x, dhi.y, dhi.z, dhi.w };

#pragma unroll
        for (int nt = 0; nt < 4; ++nt) {
            v8f acc;
#pragma unroll
            for (int v = 0; v < 8; ++v) acc[v] = bia2[nt];
            acc = __builtin_amdgcn_wmma_f32_16x16x32_f16(
                      false, A2a, false, B2r[0][nt], (short)0, acc, false, false);
            acc = __builtin_amdgcn_wmma_f32_16x16x32_f16(
                      false, A2b, false, B2r[1][nt], (short)0, acc, false, false);
#pragma unroll
            for (int v = 0; v < 8; ++v) {
                float ov = silu2 ? silu_from_half(acc[v]) : acc[v];
                atomicAdd(outAccum + (size_t)dsts[v] * HID + nt * 16 + col, ov);
            }
        }
    }
}

// h[node] = concat(x, pe)[node] @ embed_w + embed_b   (tiny: 0.36 GFLOP, VALU)
__global__ void __launch_bounds__(64)
k_embed(const float* __restrict__ x, const float* __restrict__ pe,
        const float* __restrict__ W, const float* __restrict__ b,
        float* __restrict__ h)
{
    __shared__ float f[IN_CH + PE_DIM];
    const int node = blockIdx.x;
    const int o = threadIdx.x;
    if (o < IN_CH)                 f[o] = x[(size_t)node * IN_CH + o];
    else if (o < IN_CH + PE_DIM)   f[o] = pe[(size_t)node * PE_DIM + (o - IN_CH)];
    __syncthreads();
    float acc = b[o];
#pragma unroll 8
    for (int k = 0; k < IN_CH + PE_DIM; ++k)
        acc = fmaf(f[k], W[k * HID + o], acc);
    h[(size_t)node * HID + o] = acc;
}

// h = 2h + aggr ; aggr = 0 (re-zero for next layer's scatter), float4-wide
__global__ void k_update4(float4* __restrict__ h, float4* __restrict__ aggr, int n4) {
    int i = blockIdx.x * blockDim.x + threadIdx.x;
    if (i < n4) {
        float4 hv = h[i];
        float4 av = aggr[i];
        hv.x = 2.0f * hv.x + av.x;
        hv.y = 2.0f * hv.y + av.y;
        hv.z = 2.0f * hv.z + av.z;
        hv.w = 2.0f * hv.w + av.w;
        h[i] = hv;
        aggr[i] = make_float4(0.f, 0.f, 0.f, 0.f);
    }
}

__global__ void k_zero4(float4* __restrict__ p, int n4) {
    int i = blockIdx.x * blockDim.x + threadIdx.x;
    if (i < n4) p[i] = make_float4(0.f, 0.f, 0.f, 0.f);
}

// out[g] = silu(gp[g] @ ro_w1 + ro_b1) @ ro_w2 + ro_b2   (256x64 -> 256)
__global__ void __launch_bounds__(256)
k_readout(const float* __restrict__ gp,
          const float* __restrict__ w1, const float* __restrict__ b1,
          const float* __restrict__ w2, const float* __restrict__ b2,
          float* __restrict__ out)
{
    const int g = threadIdx.x;
    float row[HID];
#pragma unroll
    for (int k = 0; k < HID; ++k) row[k] = gp[(size_t)g * HID + k];
    float acc = b2[0];
    for (int j = 0; j < HID; ++j) {
        float t = b1[j];
#pragma unroll 8
        for (int k = 0; k < HID; ++k) t = fmaf(row[k], w1[k * HID + j], t);
        acc = fmaf(silu_f(t), w2[j], acc);
    }
    out[g] = acc;
}

extern "C" void kernel_launch(void* const* d_in, const int* in_sizes, int n_in,
                              void* d_out, int out_size, void* d_ws, size_t ws_size,
                              hipStream_t stream)
{
    (void)in_sizes; (void)n_in; (void)out_size; (void)ws_size;

    const float* x       = (const float*)d_in[0];
    /* d_in[1] = pos, unused by the reference */
    const float* pe      = (const float*)d_in[2];
    const int*   eidx    = (const int*)d_in[3];
    const int*   batch   = (const int*)d_in[4];
    const float* embed_w = (const float*)d_in[5];
    const float* embed_b = (const float*)d_in[6];
    const float* msg_w1  = (const float*)d_in[7];
    const float* msg_b1  = (const float*)d_in[8];
    const float* msg_w2  = (const float*)d_in[9];
    const float* msg_b2  = (const float*)d_in[10];
    const float* pre_w1  = (const float*)d_in[11];
    const float* pre_b1  = (const float*)d_in[12];
    const float* pre_w2  = (const float*)d_in[13];
    const float* pre_b2  = (const float*)d_in[14];
    const float* ro_w1   = (const float*)d_in[15];
    const float* ro_b1   = (const float*)d_in[16];
    const float* ro_w2   = (const float*)d_in[17];
    const float* ro_b2   = (const float*)d_in[18];

    const int* send = eidx;
    const int* rec  = eidx + NN_EDGES;

    float* h    = (float*)d_ws;                       // 12.8 MB (L2-resident)
    float* aggr = h + (size_t)NN_NODES * HID;         // 12.8 MB (L2-resident)
    float* gp   = aggr + (size_t)NN_NODES * HID;      // 64 KB
    float* out  = (float*)d_out;

    const int HN4    = NN_NODES * HID / 4;
    const int zgrid  = (HN4 + 255) / 256;
    const int eTiles = NN_EDGES / 16;   // 50000 exact
    const int nTiles = NN_NODES / 16;   // 3125 exact

    k_embed<<<NN_NODES, 64, 0, stream>>>(x, pe, embed_w, embed_b, h);
    k_zero4<<<zgrid, 256, 0, stream>>>((float4*)aggr, HN4);

    for (int l = 0; l < NLAYERS; ++l) {
        k_mlp2_scatter<<<1024, 256, 0, stream>>>(
            h, send, rec,
            msg_w1 + (size_t)l * HID * HID, msg_b1 + (size_t)l * HID,
            msg_w2 + (size_t)l * HID * HID, msg_b2 + (size_t)l * HID,
            aggr, eTiles, /*silu2=*/1);
        k_update4<<<zgrid, 256, 0, stream>>>((float4*)h, (float4*)aggr, HN4);
    }

    k_zero4<<<(NN_GRAPHS * HID / 4 + 255) / 256, 256, 0, stream>>>(
        (float4*)gp, NN_GRAPHS * HID / 4);
    k_mlp2_scatter<<<256, 256, 0, stream>>>(
        h, /*gather=*/nullptr, batch,
        pre_w1, pre_b1, pre_w2, pre_b2, gp, nTiles, /*silu2=*/0);
    k_readout<<<1, 256, 0, stream>>>(gp, ro_w1, ro_b1, ro_w2, ro_b2, out);
}
